// Transformer_84104049590412
// MI455X (gfx1250) — compile-verified, matrix operand order checked by reading, not run on hardware
//
#include <hip/hip_runtime.h>

typedef _Float16 f16;
typedef __attribute__((ext_vector_type(16))) _Float16 v16h;
typedef __attribute__((ext_vector_type(8)))  _Float16 v8h;
typedef __attribute__((ext_vector_type(8)))  float    v8f;
typedef __attribute__((ext_vector_type(4)))  float    v4f;

#define D_MODEL  512
#define N_HEADS  8
#define HEAD_DIM 64
#define H_DIM    2048
#define NLAYERS  4
#define N_EMB    128
#define SRC_LEN  64
#define BATCH    8
#define MAX_ITER 12

// ---------------- f32 -> f16 weight conversion (one pass; f16 stays hot in 192MB L2) ----
__global__ void cvt_f16_kernel(const float* __restrict__ src, f16* __restrict__ dst, int n) {
  int i = blockIdx.x * blockDim.x + threadIdx.x;
  if (i < n) dst[i] = (f16)src[i];
}

// ---------------- WMMA GEMM: out[M,N] = A[M,K] * W[N,K]^T + bias (+resid)(+relu) -------
// One wave per 16x16 tile; V_WMMA_F32_16X16X32_F16, f32 accumulation.
// Loads are UNGUARDED: callers guarantee buffers have >= ceil(M/16)*16 addressable rows
// and N is tiled exactly by the grid; junk rows only feed output rows masked at the store.
__global__ __launch_bounds__(128) void gemm_wmma_kernel(
    const float* __restrict__ A, int lda,
    const f16* __restrict__ W,
    const float* __restrict__ bias,
    const float* __restrict__ resid, int ldr,
    float* __restrict__ out, int ldo,
    int M, int N, int K, int relu)
{
  const int wave = threadIdx.x >> 5;
  const int lane = threadIdx.x & 31;
  const int half = lane >> 4;          // CDNA5 16-bit operand layout: hi lanes hold K+8
  const int r    = lane & 15;
  const int m_base = blockIdx.x * 16;
  const int n_base = (blockIdx.y * 4 + wave) * 16;
  if (n_base >= N) return;

  const float* Ap = A + (size_t)(m_base + r) * lda + 8 * half;
  const f16*   Wp = W + (size_t)(n_base + r) * K   + 8 * half;

  v8f c = {};
  for (int k0 = 0; k0 < K; k0 += 32) {
    // A fragment: elements 0-7 = K[8h .. 8h+7], elements 8-15 = K[16+8h .. 16+8h+7]
    const v4f a0 = *(const v4f*)(Ap + k0);
    const v4f a1 = *(const v4f*)(Ap + k0 + 4);
    const v4f a2 = *(const v4f*)(Ap + k0 + 16);
    const v4f a3 = *(const v4f*)(Ap + k0 + 20);
    // B fragment: same K runs, already f16 -> two 16-byte loads
    const v8h b0 = *(const v8h*)(Wp + k0);
    const v8h b1 = *(const v8h*)(Wp + k0 + 16);

    v16h a, b;
#pragma unroll
    for (int e = 0; e < 4; ++e) {
      a[e]      = (f16)a0[e];
      a[e + 4]  = (f16)a1[e];
      a[e + 8]  = (f16)a2[e];
      a[e + 12] = (f16)a3[e];
    }
#pragma unroll
    for (int e = 0; e < 8; ++e) { b[e] = b0[e]; b[e + 8] = b1[e]; }

    c = __builtin_amdgcn_wmma_f32_16x16x32_f16(false, a, false, b, (short)0, c, false, false);
  }

  const int col = n_base + r;
  const float bb = bias ? bias[col] : 0.f;
#pragma unroll
  for (int v = 0; v < 8; ++v) {
    const int row = m_base + v + 8 * half;   // C/D layout: VGPR v -> M = v (+8 for hi lanes)
    if (row < M) {
      float val = c[v] + bb;
      if (resid) val += resid[(size_t)row * ldr + col];
      if (relu)  val = fmaxf(val, 0.f);
      out[(size_t)row * ldo + col] = val;
    }
  }
}

// ---------------- attention: one thread per query row, per (batch, head) --------------
__global__ __launch_bounds__(64) void attn_kernel(
    const float* __restrict__ Q, int ldq, int qoff,
    const float* __restrict__ Km, int ldk, int koff,
    const float* __restrict__ Vm, int ldv, int voff,
    float* __restrict__ O, int ldo, int Sq, int Sk)
{
  const int b = blockIdx.x >> 3;
  const int h = blockIdx.x & 7;
  const int s = threadIdx.x;
  if (s >= Sq) return;
  const float* q = Q + ((size_t)s * BATCH + b) * ldq + qoff + h * HEAD_DIM;
  float qr[HEAD_DIM];
#pragma unroll
  for (int d = 0; d < HEAD_DIM; ++d) qr[d] = q[d] * 0.125f;   // 1/sqrt(64)
  float sc[SRC_LEN];
  float mx = -3.0e38f;
  for (int k = 0; k < Sk; ++k) {
    const float* kr = Km + ((size_t)k * BATCH + b) * ldk + koff + h * HEAD_DIM;
    float dot = 0.f;
#pragma unroll
    for (int d = 0; d < HEAD_DIM; ++d) dot += qr[d] * kr[d];
    sc[k] = dot;
    mx = fmaxf(mx, dot);
  }
  float sum = 0.f;
  for (int k = 0; k < Sk; ++k) { const float e = expf(sc[k] - mx); sc[k] = e; sum += e; }
  const float inv = 1.f / sum;
  float acc[HEAD_DIM];
#pragma unroll
  for (int d = 0; d < HEAD_DIM; ++d) acc[d] = 0.f;
  for (int k = 0; k < Sk; ++k) {
    const float w = sc[k] * inv;
    const float* vr = Vm + ((size_t)k * BATCH + b) * ldv + voff + h * HEAD_DIM;
#pragma unroll
    for (int d = 0; d < HEAD_DIM; ++d) acc[d] += w * vr[d];
  }
  float* o = O + ((size_t)s * BATCH + b) * ldo + h * HEAD_DIM;
#pragma unroll
  for (int d = 0; d < HEAD_DIM; ++d) o[d] = acc[d];
}

// ---------------- in-place LayerNorm, one wave32 per row of 512 ------------------------
__global__ __launch_bounds__(32) void ln_kernel(float* __restrict__ X,
                                                const float* __restrict__ g,
                                                const float* __restrict__ bta) {
  const int row = blockIdx.x;
  const int lane = threadIdx.x;
  float r[16];
  float s = 0.f;
#pragma unroll
  for (int j = 0; j < 16; ++j) { r[j] = X[(size_t)row * D_MODEL + lane + j * 32]; s += r[j]; }
#pragma unroll
  for (int off = 16; off > 0; off >>= 1) s += __shfl_xor(s, off, 32);
  const float mean = s * (1.f / D_MODEL);
  float v = 0.f;
#pragma unroll
  for (int j = 0; j < 16; ++j) { const float d = r[j] - mean; v += d * d; }
#pragma unroll
  for (int off = 16; off > 0; off >>= 1) v += __shfl_xor(v, off, 32);
  const float inv = rsqrtf(v * (1.f / D_MODEL) + 1e-5f);
#pragma unroll
  for (int j = 0; j < 16; ++j) {
    const int c = lane + j * 32;
    X[(size_t)row * D_MODEL + c] = g[c] * (r[j] - mean) * inv + bta[c];
  }
}

// ---------------- embeddings ----------------------------------------------------------
__global__ void embed_enc_kernel(const int* __restrict__ x, const float* __restrict__ embW,
                                 float* __restrict__ mem) {
  const int row = blockIdx.x;            // s*BATCH + b
  const int s = row >> 3;
  const int tok = x[row];
  for (int c = threadIdx.x; c < D_MODEL; c += blockDim.x) {
    const int i2 = c & ~1;
    const float div = expf((float)i2 * (-9.210340371976184f / (float)D_MODEL));
    const float ang = (float)s * div;
    const float pe = (c & 1) ? cosf(ang) : sinf(ang);
    mem[(size_t)row * D_MODEL + c] = embW[(size_t)tok * D_MODEL + c] + pe;
  }
}

__global__ void embed_dec_kernel(const int* __restrict__ curr, const float* __restrict__ embW,
                                 float* __restrict__ t) {
  const int row = blockIdx.x;
  const int tok = curr[row];
  for (int c = threadIdx.x; c < D_MODEL; c += blockDim.x)
    t[(size_t)row * D_MODEL + c] = embW[(size_t)tok * D_MODEL + c];
}

__global__ void init_curr_kernel(int* __restrict__ curr, const int* __restrict__ bos) {
  if (threadIdx.x < BATCH) curr[threadIdx.x] = *bos;
}

// ---------------- greedy argmax over 128 logits, wave32 per batch element --------------
__global__ __launch_bounds__(32) void argmax_kernel(const float* __restrict__ logits,
                                                    int* __restrict__ next) {
  const int b = blockIdx.x;
  const int lane = threadIdx.x;
  float best = -3.0e38f; int bi = 0x7fffffff;
#pragma unroll
  for (int j = 0; j < 4; ++j) {
    const int idx = lane + j * 32;
    const float v = logits[b * N_EMB + idx];
    if (v > best) { best = v; bi = idx; }        // ascending idx + strict > => lowest index on tie
  }
#pragma unroll
  for (int off = 16; off > 0; off >>= 1) {
    const float ov = __shfl_xor(best, off, 32);
    const int   oi = __shfl_xor(bi, off, 32);
    if (ov > best || (ov == best && oi < bi)) { best = ov; bi = oi; }
  }
  if (lane == 0) next[b] = bi;
}

// ---------------- host orchestration --------------------------------------------------
static inline void launch_gemm(hipStream_t st, const float* A, int lda, const f16* W,
                               const float* bias, const float* resid, int ldr,
                               float* out, int ldo, int M, int N, int K, int relu) {
  dim3 g((M + 15) / 16, (N + 63) / 64);
  gemm_wmma_kernel<<<g, 128, 0, st>>>(A, lda, W, bias, resid, ldr, out, ldo, M, N, K, relu);
}

extern "C" void kernel_launch(void* const* d_in, const int* in_sizes, int n_in,
                              void* d_out, int out_size, void* d_ws, size_t ws_size,
                              hipStream_t stream) {
  (void)in_sizes; (void)n_in; (void)out_size; (void)ws_size;
  int ix = 0;
  const int*   d_x    = (const int*)  d_in[ix++];
  const float* d_embW = (const float*)d_in[ix++];

  struct AttnP { const float *in_w, *in_b, *out_w, *out_b; };
  struct EncP { AttnP sa; const float *ln1g,*ln1b,*ln2g,*ln2b,*w1,*b1,*w2,*b2; };
  struct DecP { AttnP sa, ca; const float *ln1g,*ln1b,*ln2g,*ln2b,*ln3g,*ln3b,*w1,*b1,*w2,*b2; };
  EncP enc[NLAYERS]; DecP dec[NLAYERS];
  auto rdF = [&]() { return (const float*)d_in[ix++]; };
  auto rdA = [&](AttnP& a) { a.in_w=rdF(); a.in_b=rdF(); a.out_w=rdF(); a.out_b=rdF(); };
  for (int l = 0; l < NLAYERS; ++l) {
    rdA(enc[l].sa);
    enc[l].ln1g=rdF(); enc[l].ln1b=rdF(); enc[l].ln2g=rdF(); enc[l].ln2b=rdF();
    enc[l].w1=rdF(); enc[l].b1=rdF(); enc[l].w2=rdF(); enc[l].b2=rdF();
  }
  for (int l = 0; l < NLAYERS; ++l) {
    rdA(dec[l].sa); rdA(dec[l].ca);
    dec[l].ln1g=rdF(); dec[l].ln1b=rdF(); dec[l].ln2g=rdF(); dec[l].ln2b=rdF();
    dec[l].ln3g=rdF(); dec[l].ln3b=rdF();
    dec[l].w1=rdF(); dec[l].b1=rdF(); dec[l].w2=rdF(); dec[l].b2=rdF();
  }
  const float* d_outW = rdF();
  const float* d_outb = rdF();
  ix++;                                            // max_iter (== 12, compile-time in ref)
  const int* d_bos = (const int*)d_in[ix++];

  // ---- workspace carve ----
  size_t off = 0;
  auto carve = [&](size_t bytes) -> void* {
    void* p = (char*)d_ws + off;
    off += (bytes + 255) & ~(size_t)255;
    return p;
  };
  f16 *enc_inw[NLAYERS], *enc_outw[NLAYERS], *enc_w1h[NLAYERS], *enc_w2h[NLAYERS];
  f16 *dec_sainw[NLAYERS], *dec_saoutw[NLAYERS], *dec_cainw[NLAYERS], *dec_caoutw[NLAYERS];
  f16 *dec_w1h[NLAYERS], *dec_w2h[NLAYERS];
  for (int l = 0; l < NLAYERS; ++l) {
    enc_inw[l]  = (f16*)carve((size_t)3*D_MODEL*D_MODEL*2);
    enc_outw[l] = (f16*)carve((size_t)D_MODEL*D_MODEL*2);
    enc_w1h[l]  = (f16*)carve((size_t)H_DIM*D_MODEL*2);
    enc_w2h[l]  = (f16*)carve((size_t)D_MODEL*H_DIM*2);
  }
  for (int l = 0; l < NLAYERS; ++l) {
    dec_sainw[l]  = (f16*)carve((size_t)3*D_MODEL*D_MODEL*2);
    dec_saoutw[l] = (f16*)carve((size_t)D_MODEL*D_MODEL*2);
    dec_cainw[l]  = (f16*)carve((size_t)3*D_MODEL*D_MODEL*2);
    dec_caoutw[l] = (f16*)carve((size_t)D_MODEL*D_MODEL*2);
    dec_w1h[l]    = (f16*)carve((size_t)H_DIM*D_MODEL*2);
    dec_w2h[l]    = (f16*)carve((size_t)D_MODEL*H_DIM*2);
  }
  f16* outW_h = (f16*)carve((size_t)N_EMB*D_MODEL*2);

  float* mem     = (float*)carve((size_t)512*512*4);
  float* bufBig  = (float*)carve((size_t)512*2048*4);
  float* bufAtt  = (float*)carve((size_t)512*512*4);
  float* bufT2   = (float*)carve((size_t)512*512*4);
  float* kvmem[NLAYERS];
  for (int l = 0; l < NLAYERS; ++l) kvmem[l] = (float*)carve((size_t)512*1024*4);
  float* tA      = (float*)carve((size_t)96*512*4);
  float* tB      = (float*)carve((size_t)96*512*4);
  float* qbuf    = (float*)carve((size_t)96*512*4);
  float* logits8 = (float*)carve((size_t)16*N_EMB*4);     // 16 rows: full store-tile slack
  int*   curr    = (int*)  carve((size_t)(MAX_ITER+1)*BATCH*4);

  // ---- one-time weight conversion to f16 (resident in L2 thereafter) ----
  auto cvt = [&](const float* s, f16* d, int n) {
    cvt_f16_kernel<<<(n + 255) / 256, 256, 0, stream>>>(s, d, n);
  };
  for (int l = 0; l < NLAYERS; ++l) {
    cvt(enc[l].sa.in_w,  enc_inw[l],  3*D_MODEL*D_MODEL);
    cvt(enc[l].sa.out_w, enc_outw[l], D_MODEL*D_MODEL);
    cvt(enc[l].w1,       enc_w1h[l],  H_DIM*D_MODEL);
    cvt(enc[l].w2,       enc_w2h[l],  D_MODEL*H_DIM);
  }
  for (int l = 0; l < NLAYERS; ++l) {
    cvt(dec[l].sa.in_w,  dec_sainw[l],  3*D_MODEL*D_MODEL);
    cvt(dec[l].sa.out_w, dec_saoutw[l], D_MODEL*D_MODEL);
    cvt(dec[l].ca.in_w,  dec_cainw[l],  3*D_MODEL*D_MODEL);
    cvt(dec[l].ca.out_w, dec_caoutw[l], D_MODEL*D_MODEL);
    cvt(dec[l].w1,       dec_w1h[l],    H_DIM*D_MODEL);
    cvt(dec[l].w2,       dec_w2h[l],    D_MODEL*H_DIM);
  }
  cvt(d_outW, outW_h, N_EMB*D_MODEL);

  // ---- encoder ----
  const int Me = SRC_LEN * BATCH;   // 512
  embed_enc_kernel<<<Me, 256, 0, stream>>>(d_x, d_embW, mem);
  for (int l = 0; l < NLAYERS; ++l) {
    launch_gemm(stream, mem, 512, enc_inw[l], enc[l].sa.in_b, nullptr, 0, bufBig, 1536, Me, 1536, 512, 0);
    attn_kernel<<<BATCH*N_HEADS, 64, 0, stream>>>(bufBig,1536,0, bufBig,1536,512, bufBig,1536,1024,
                                                  bufAtt,512, SRC_LEN, SRC_LEN);
    launch_gemm(stream, bufAtt, 512, enc_outw[l], enc[l].sa.out_b, mem, 512, bufT2, 512, Me, 512, 512, 0);
    ln_kernel<<<Me, 32, 0, stream>>>(bufT2, enc[l].ln1g, enc[l].ln1b);
    launch_gemm(stream, bufT2, 512, enc_w1h[l], enc[l].b1, nullptr, 0, bufBig, 2048, Me, 2048, 512, 1);
    launch_gemm(stream, bufBig, 2048, enc_w2h[l], enc[l].b2, bufT2, 512, mem, 512, Me, 512, 2048, 0);
    ln_kernel<<<Me, 32, 0, stream>>>(mem, enc[l].ln2g, enc[l].ln2b);
  }

  // ---- hoist loop-invariant cross-attention K/V out of the decode loop ----
  for (int l = 0; l < NLAYERS; ++l)
    launch_gemm(stream, mem, 512, dec_cainw[l] + (size_t)512*512, dec[l].ca.in_b + 512,
                nullptr, 0, kvmem[l], 1024, Me, 1024, 512, 0);

  // ---- greedy decode ----
  init_curr_kernel<<<1, 32, 0, stream>>>(curr, d_bos);
  for (int it = 0; it < MAX_ITER; ++it) {
    const int rows = (it + 1) * BATCH;
    float* t0 = tA; float* t1 = tB;
    embed_dec_kernel<<<rows, 256, 0, stream>>>(curr, d_embW, t0);
    for (int l = 0; l < NLAYERS; ++l) {
      launch_gemm(stream, t0, 512, dec_sainw[l], dec[l].sa.in_b, nullptr, 0, bufBig, 1536, rows, 1536, 512, 0);
      attn_kernel<<<BATCH*N_HEADS, 64, 0, stream>>>(bufBig,1536,0, bufBig,1536,512, bufBig,1536,1024,
                                                    bufAtt,512, it+1, it+1);
      launch_gemm(stream, bufAtt, 512, dec_saoutw[l], dec[l].sa.out_b, t0, 512, t1, 512, rows, 512, 512, 0);
      ln_kernel<<<rows, 32, 0, stream>>>(t1, dec[l].ln1g, dec[l].ln1b);
      launch_gemm(stream, t1, 512, dec_cainw[l], dec[l].ca.in_b, nullptr, 0, qbuf, 512, rows, 512, 512, 0);
      attn_kernel<<<BATCH*N_HEADS, 64, 0, stream>>>(qbuf,512,0, kvmem[l],1024,0, kvmem[l],1024,512,
                                                    bufAtt,512, it+1, SRC_LEN);
      launch_gemm(stream, bufAtt, 512, dec_caoutw[l], dec[l].ca.out_b, t1, 512, t0, 512, rows, 512, 512, 0);
      ln_kernel<<<rows, 32, 0, stream>>>(t0, dec[l].ln2g, dec[l].ln2b);
      launch_gemm(stream, t0, 512, dec_w1h[l], dec[l].b1, nullptr, 0, bufBig, 2048, rows, 2048, 512, 1);
      launch_gemm(stream, bufBig, 2048, dec_w2h[l], dec[l].b2, t0, 512, t1, 512, rows, 512, 2048, 0);
      ln_kernel<<<rows, 32, 0, stream>>>(t1, dec[l].ln3g, dec[l].ln3b);
      float* tmp = t0; t0 = t1; t1 = tmp;          // layer output now in t0
    }
    if (it == MAX_ITER - 1) {
      // final iteration: full logits straight to output buffer
      launch_gemm(stream, t0, 512, outW_h, d_outb, nullptr, 0, (float*)d_out, N_EMB, rows, N_EMB, 512, 0);
    } else {
      // only the last token's logits feed argmax
      launch_gemm(stream, t0 + (size_t)it * BATCH * 512, 512, outW_h, d_outb, nullptr, 0,
                  logits8, N_EMB, BATCH, N_EMB, 512, 0);
      argmax_kernel<<<BATCH, 32, 0, stream>>>(logits8, curr + (size_t)(it + 1) * BATCH);
    }
  }
}
